// Transformer_38079180046466
// MI455X (gfx1250) — matrix operand order repaired and checked by reading, not compiled, not run on hardware
//
#include <hip/hip_runtime.h>
#include <hip/hip_bf16.h>
#include <math.h>
#include <stdint.h>

// ---------------------------------------------------------------------------
// ViT forward on gfx1250 (MI455X).
// GEMMs: v_wmma_f32_16x16x32_f16 fed by ds_load_b128 from f16 LDS tiles,
// staged with global_load_async_to_lds_b128 (ASYNCcnt), double-buffered.
// Weights are pre-converted (per launch) to f16 transposed [N][K] so both
// A and B fragments are K-contiguous 16B LDS loads -- no cvt in the hot loop.
// ---------------------------------------------------------------------------

typedef __attribute__((ext_vector_type(16))) _Float16 v16h;
typedef __attribute__((ext_vector_type(8)))  _Float16 v8h;
typedef __attribute__((ext_vector_type(8)))  float    v8f;

#define MROWS   528     // 33 * 16 (padded)
#define MVALID  520     // 8 * 65
#define EDIM    256
#define HEADS   2
#define HD      128
#define NTOK    65
#define GROUPS  9
#define BS      8
#define FF      1024
#define LAYERS  10

// ---------------------------------------------------------------------------
// CDNA5 async copy: global -> LDS, 16 bytes per lane, tracked by ASYNCcnt.
// ---------------------------------------------------------------------------
__device__ __forceinline__ void async_cp_b128(uint32_t lds_byte_off, const void* gptr) {
    asm volatile("global_load_async_to_lds_b128 %0, %1, off"
                 :: "v"(lds_byte_off), "v"((uint64_t)(uintptr_t)gptr)
                 : "memory");
}
__device__ __forceinline__ uint32_t lds_off(const void* p) {
    return (uint32_t)(uintptr_t)p;   // low 32 bits of generic LDS ptr = offset
}

// ---------------------------------------------------------------------------
// Weight convert + transpose: src f32 row-major KxN (per layer) ->
// dst f16 [N][K]. grid.y = layer.
// ---------------------------------------------------------------------------
__global__ void convert_w_kernel(const float* __restrict__ src,
                                 _Float16* __restrict__ dst, int K, int N) {
    size_t base = (size_t)blockIdx.y * K * N;
    int idx = blockIdx.x * 256 + threadIdx.x;
    if (idx >= K * N) return;
    int n = idx / K, kk = idx % K;
    dst[base + idx] = (_Float16)src[base + (size_t)kk * N + n];
}

// ---------------------------------------------------------------------------
// Patch embedding (h stays f32; pad rows zeroed).
// ---------------------------------------------------------------------------
__global__ void patch_embed_kernel(const float* __restrict__ src,
                                   const float* __restrict__ conv_w,
                                   const float* __restrict__ conv_b,
                                   const float* __restrict__ pos,
                                   const float* __restrict__ cls,
                                   float* __restrict__ h) {
    int idx = blockIdx.x * blockDim.x + threadIdx.x;
    if (idx >= MROWS * EDIM) return;
    int row = idx >> 8;
    int e   = idx & 255;
    if (row >= MVALID) { h[idx] = 0.0f; return; }
    int b = row / NTOK;
    int t = row % NTOK;
    float val;
    if (t == 0) {
        val = cls[e] + pos[e];
    } else {
        int p  = t - 1;
        int py = p >> 3, px = p & 7;
        float acc = conv_b[e];
        #pragma unroll
        for (int c = 0; c < 3; ++c)
            #pragma unroll
            for (int i = 0; i < 8; ++i)
                #pragma unroll
                for (int j = 0; j < 8; ++j)
                    acc += src[((b * 3 + c) * 64 + (py * 8 + i)) * 64 + (px * 8 + j)]
                         * conv_w[((e * 3 + c) * 8 + i) * 8 + j];
        val = acc + pos[t * EDIM + e];
    }
    h[idx] = val;
}

// ---------------------------------------------------------------------------
// LayerNorm: f32 in, f16 out (feeds WMMA A); pad rows zeroed.
// ---------------------------------------------------------------------------
__global__ void ln_kernel(const float* __restrict__ X,
                          const float* __restrict__ g,
                          const float* __restrict__ b,
                          _Float16* __restrict__ Y) {
    int row = blockIdx.x;
    int tid = threadIdx.x;
    if (row >= MVALID) { Y[row * EDIM + tid] = (_Float16)0.0f; return; }
    __shared__ float red[EDIM];
    float x = X[row * EDIM + tid];
    red[tid] = x;
    __syncthreads();
    for (int s = 128; s > 0; s >>= 1) {
        if (tid < s) red[tid] += red[tid + s];
        __syncthreads();
    }
    float mean = red[0] * (1.0f / EDIM);
    __syncthreads();
    float d = x - mean;
    red[tid] = d * d;
    __syncthreads();
    for (int s = 128; s > 0; s >>= 1) {
        if (tid < s) red[tid] += red[tid + s];
        __syncthreads();
    }
    float var = red[0] * (1.0f / EDIM);
    Y[row * EDIM + tid] = (_Float16)(d * rsqrtf(var + 1e-5f) * g[tid] + b[tid]);
}

// ---------------------------------------------------------------------------
// WMMA GEMM, f16 end-to-end data path, async-LDS staged, double buffered.
//   Block = 128 threads = 4 waves; one 16(M) x 64(N) tile; grid = (N/64, 33).
//   Per K-chunk (CK=64): As[16][64] f16 (2KB) + Wst[64][64] f16 (8KB) staged
//   with exactly 5 global_load_async_to_lds_b128 per thread (uniform per
//   wave) -> s_wait_asynccnt 0x5 drains the previous chunk only.
//   Fragments are pure ds_load_b128 pairs (A and B both K-contiguous):
//     A: af[0..7] <-> k=khalf*8+0..7, af[8..15] <-> k=16+khalf*8+0..7
//     B (from transposed W): bf[0..7] <-> k=khalf*8+0..7, bf[8..15] <-> k=16+khalf*8+0..7
//     C/D f32: lane n=l%16, vgpr v -> M = v + 8*(l/16)
// ---------------------------------------------------------------------------
template<int KDIM, bool GELU, bool ACCUM, typename OutT>
__global__ __launch_bounds__(128)
void gemm_wmma_kernel(const _Float16* __restrict__ A, int lda,
                      const _Float16* __restrict__ Wt,   // [N][KDIM] f16
                      const float* __restrict__ bias,
                      OutT* __restrict__ C, int ldc) {
    constexpr int CK = 64;
    __shared__ __align__(16) _Float16 As [2][16][CK];   //  4 KB
    __shared__ __align__(16) _Float16 Wst[2][64][CK];   // 16 KB

    const int tid  = threadIdx.x;
    const int wave = tid >> 5;
    const int lane = tid & 31;
    const int mt   = blockIdx.y;        // M tile
    const int ntg  = blockIdx.x;        // 64-column N panel

    // Staging coordinates: A panel 16x64 f16 = 128 * 16B; W panel 64x64 f16.
    const int aR = tid >> 3, aC = (tid & 7) * 8;
    const _Float16* Abase = A + (size_t)(mt * 16 + aR) * lda + aC;
    const _Float16* Wbase = Wt + (size_t)ntg * 64 * KDIM;

    auto stage = [&](int buf, int kk) {
        async_cp_b128(lds_off(&As[buf][aR][aC]), Abase + kk);
        #pragma unroll
        for (int r = 0; r < 4; ++r) {
            int i2 = tid * 8 + r * 1024;          // 4096 f16 -> 4 ops/thread
            int wR = i2 >> 6, wC = i2 & 63;
            async_cp_b128(lds_off(&Wst[buf][wR][wC]),
                          Wbase + (size_t)wR * KDIM + kk + wC);
        }
    };

    const int mi    = lane & 15;
    const int khalf = lane >> 4;
    const int ncol  = wave * 16 + (lane & 15);   // column within 64-wide panel

    v8f acc = {};
    int buf = 0;
    stage(0, 0);
    #pragma unroll 2
    for (int kk = 0; kk < KDIM; kk += CK) {
        if (kk + CK < KDIM) {
            stage(buf ^ 1, kk + CK);
            asm volatile("s_wait_asynccnt 0x5" ::: "memory");  // prev chunk done
        } else {
            asm volatile("s_wait_asynccnt 0x0" ::: "memory");
        }
        __syncthreads();

        #pragma unroll
        for (int s = 0; s < 2; ++s) {            // two k=32 WMMAs per chunk
            v8h alo = *(const v8h*)&As[buf][mi][s * 32 + khalf * 8];
            v8h ahi = *(const v8h*)&As[buf][mi][s * 32 + 16 + khalf * 8];
            v16h af = __builtin_shufflevector(alo, ahi,
                0,1,2,3,4,5,6,7,8,9,10,11,12,13,14,15);
            v8h blo = *(const v8h*)&Wst[buf][ncol][s * 32 + khalf * 8];
            v8h bhi = *(const v8h*)&Wst[buf][ncol][s * 32 + 16 + khalf * 8];
            v16h bf = __builtin_shufflevector(blo, bhi,
                0,1,2,3,4,5,6,7,8,9,10,11,12,13,14,15);
            acc = __builtin_amdgcn_wmma_f32_16x16x32_f16(
                false, af, false, bf, (short)0, acc, false, false);
        }

        __syncthreads();            // all waves done reading buf before restage
        buf ^= 1;
    }

    const int n     = lane & 15;
    const int mbase = (lane >> 4) * 8;
    const int ocol  = ntg * 64 + wave * 16 + n;
    const float bval = bias ? bias[ocol] : 0.0f;
    #pragma unroll
    for (int v = 0; v < 8; ++v) {
        int row = mt * 16 + mbase + v;
        float x = acc[v] + bval;
        if (GELU) x = 0.5f * x * (1.0f + erff(x * 0.70710678118654752f));
        OutT* p = C + (size_t)row * ldc + ocol;
        if (ACCUM) *p = (OutT)((float)*p + x);
        else       *p = (OutT)x;
    }
}

// ---------------------------------------------------------------------------
// Block-diagonal attention (BS=8) with fused residual into h (all f32).
// ---------------------------------------------------------------------------
__global__ void attn_kernel(const float* __restrict__ qp,
                            const float* __restrict__ kp,
                            const float* __restrict__ vv,
                            float* __restrict__ h) {
    const int bid  = blockIdx.x;            // 8*9*2 = 144
    const int head = bid % HEADS;
    const int g    = (bid / HEADS) % GROUPS;
    const int b    = bid / (HEADS * GROUPS);
    const int tid  = threadIdx.x;           // 64
    const float inv_scale = 1.0f / 11.313708498984761f;  // 1/sqrt(HD)

    __shared__ float sc[BS][BS];
    __shared__ float w [BS][BS];

    const int qi = tid >> 3, ki = tid & 7;
    const int tq = g * BS + qi, tk = g * BS + ki;
    const int rowq = b * NTOK + tq;         // <= 526, inside 528-row buffers
    const int rowk = b * NTOK + tk;

    float s;
    if (tk < NTOK) {
        const float* qr = qp + (size_t)rowq * EDIM + head * HD;
        const float* kr = kp + (size_t)rowk * EDIM + head * HD;
        float acc = 0.0f;
        #pragma unroll 8
        for (int d = 0; d < HD; ++d) acc += qr[d] * kr[d];
        s = acc * inv_scale;
    } else {
        s = -1e30f;                         // masked (padded key)
    }
    sc[qi][ki] = s;
    __syncthreads();

    float mx = sc[qi][0];
    #pragma unroll
    for (int j = 1; j < BS; ++j) mx = fmaxf(mx, sc[qi][j]);
    float sum = 0.0f;
    #pragma unroll
    for (int j = 0; j < BS; ++j) sum += expf(sc[qi][j] - mx);
    w[qi][ki] = expf(s - mx) / sum;
    __syncthreads();

    for (int o = tid; o < BS * HD; o += 64) {
        int q2 = o >> 7, d = o & (HD - 1);
        int t2 = g * BS + q2;
        if (t2 >= NTOK) continue;
        float acc = 0.0f;
        #pragma unroll
        for (int k2 = 0; k2 < BS; ++k2)
            acc += w[q2][k2] * vv[(size_t)(b * NTOK + g * BS + k2) * EDIM + head * HD + d];
        h[(size_t)(b * NTOK + t2) * EDIM + head * HD + d] += acc;
    }
}

// ---------------------------------------------------------------------------
// Classifier head: out[b, c] = h[b*65] . fc_w[:, c] + fc_b[c]   (8x10)
// ---------------------------------------------------------------------------
__global__ void head_kernel(const float* __restrict__ h,
                            const float* __restrict__ fcw,
                            const float* __restrict__ fcb,
                            float* __restrict__ out) {
    int tid = threadIdx.x;
    if (tid >= 80) return;
    int b = tid / 10, c = tid % 10;
    const float* x = h + (size_t)(b * NTOK) * EDIM;
    float acc = fcb[c];
    for (int k = 0; k < EDIM; ++k) acc += x[k] * fcw[k * 10 + c];
    out[tid] = acc;
}

// ---------------------------------------------------------------------------
extern "C" void kernel_launch(void* const* d_in, const int* in_sizes, int n_in,
                              void* d_out, int out_size, void* d_ws, size_t ws_size,
                              hipStream_t stream) {
    const float* src    = (const float*)d_in[0];
    const float* conv_w = (const float*)d_in[1];
    const float* conv_b = (const float*)d_in[2];
    const float* pos    = (const float*)d_in[3];
    const float* cls    = (const float*)d_in[4];
    const float* Wq     = (const float*)d_in[5];
    const float* bq     = (const float*)d_in[6];
    const float* Wk     = (const float*)d_in[7];
    const float* bk     = (const float*)d_in[8];
    const float* Wv     = (const float*)d_in[9];
    const float* bv     = (const float*)d_in[10];
    const float* fproj  = (const float*)d_in[11];
    const float* ln1_g  = (const float*)d_in[12];
    const float* ln1_b  = (const float*)d_in[13];
    const float* ln2_g  = (const float*)d_in[14];
    const float* ln2_b  = (const float*)d_in[15];
    const float* ff1_w  = (const float*)d_in[16];
    const float* ff1_b  = (const float*)d_in[17];
    const float* ff2_w  = (const float*)d_in[18];
    const float* ff2_b  = (const float*)d_in[19];
    const float* fc_w   = (const float*)d_in[20];
    const float* fc_b   = (const float*)d_in[21];

    // ---- workspace carve-up: f32 region first, then f16 region (~19 MB) ----
    const size_t ACT = (size_t)MROWS * EDIM;     // 135168
    float* f32ws = (float*)d_ws;
    float* h   = f32ws;  f32ws += ACT;
    float* v   = f32ws;  f32ws += ACT;
    float* qp  = f32ws;  f32ws += ACT;
    float* kp  = f32ws;  f32ws += ACT;
    _Float16* f16ws = (_Float16*)f32ws;
    _Float16* xn  = f16ws;  f16ws += ACT;
    _Float16* qh  = f16ws;  f16ws += ACT;
    _Float16* kh  = f16ws;  f16ws += ACT;
    _Float16* f1h = f16ws;  f16ws += (size_t)MROWS * FF;
    _Float16* wqT = f16ws;  f16ws += (size_t)LAYERS * EDIM * EDIM;
    _Float16* wkT = f16ws;  f16ws += (size_t)LAYERS * EDIM * EDIM;
    _Float16* wvT = f16ws;  f16ws += (size_t)LAYERS * EDIM * EDIM;
    _Float16* fpT = f16ws;  f16ws += (size_t)LAYERS * HD * HD;
    _Float16* w1T = f16ws;  f16ws += (size_t)LAYERS * EDIM * FF;
    _Float16* w2T = f16ws;  f16ws += (size_t)LAYERS * FF * EDIM;

    const int MT = MROWS / 16;                   // 33 M-tiles

    // 0) Convert + transpose all weights to f16 [N][K] (per-launch, cheap)
    convert_w_kernel<<<dim3((EDIM*EDIM+255)/256, LAYERS), 256, 0, stream>>>(Wq, wqT, EDIM, EDIM);
    convert_w_kernel<<<dim3((EDIM*EDIM+255)/256, LAYERS), 256, 0, stream>>>(Wk, wkT, EDIM, EDIM);
    convert_w_kernel<<<dim3((EDIM*EDIM+255)/256, LAYERS), 256, 0, stream>>>(Wv, wvT, EDIM, EDIM);
    convert_w_kernel<<<dim3((HD*HD+255)/256,     LAYERS), 256, 0, stream>>>(fproj, fpT, HD, HD);
    convert_w_kernel<<<dim3((EDIM*FF+255)/256,   LAYERS), 256, 0, stream>>>(ff1_w, w1T, EDIM, FF);
    convert_w_kernel<<<dim3((FF*EDIM+255)/256,   LAYERS), 256, 0, stream>>>(ff2_w, w2T, FF, EDIM);

    // 1) Patch embed + cls + pos
    patch_embed_kernel<<<(MROWS * EDIM + 255) / 256, 256, 0, stream>>>(
        src, conv_w, conv_b, pos, cls, h);

    for (int l = 0; l < LAYERS; ++l) {
        const _Float16* wqt = wqT + (size_t)l * EDIM * EDIM;
        const _Float16* wkt = wkT + (size_t)l * EDIM * EDIM;
        const _Float16* wvt = wvT + (size_t)l * EDIM * EDIM;
        const _Float16* fpt = fpT + (size_t)l * HD * HD;
        const _Float16* w1t = w1T + (size_t)l * EDIM * FF;
        const _Float16* w2t = w2T + (size_t)l * FF * EDIM;

        // LN1 -> f16
        ln_kernel<<<MROWS, EDIM, 0, stream>>>(h, ln1_g + l * EDIM, ln1_b + l * EDIM, xn);

        // QKV (K=256): q,k to f16 (feed fproj GEMM); v to f32 (feeds attn)
        gemm_wmma_kernel<EDIM, false, false, _Float16><<<dim3(4, MT), 128, 0, stream>>>(
            xn, EDIM, wqt, bq + l * EDIM, qh, EDIM);
        gemm_wmma_kernel<EDIM, false, false, _Float16><<<dim3(4, MT), 128, 0, stream>>>(
            xn, EDIM, wkt, bk + l * EDIM, kh, EDIM);
        gemm_wmma_kernel<EDIM, false, false, float><<<dim3(4, MT), 128, 0, stream>>>(
            xn, EDIM, wvt, bv + l * EDIM, v, EDIM);

        // Per-head feature projection (K=128): f16 in, f32 out
        for (int hh = 0; hh < HEADS; ++hh) {
            gemm_wmma_kernel<HD, false, false, float><<<dim3(2, MT), 128, 0, stream>>>(
                qh + hh * HD, EDIM, fpt, nullptr, qp + hh * HD, EDIM);
            gemm_wmma_kernel<HD, false, false, float><<<dim3(2, MT), 128, 0, stream>>>(
                kh + hh * HD, EDIM, fpt, nullptr, kp + hh * HD, EDIM);
        }

        // Block-diagonal attention with fused residual into h
        attn_kernel<<<8 * GROUPS * HEADS, 64, 0, stream>>>(qp, kp, v, h);

        // LN2 -> f16
        ln_kernel<<<MROWS, EDIM, 0, stream>>>(h, ln2_g + l * EDIM, ln2_b + l * EDIM, xn);

        // FF1 + exact GELU (K=256, N=1024): f16 out (feeds FF2)
        gemm_wmma_kernel<EDIM, true, false, _Float16><<<dim3(16, MT), 128, 0, stream>>>(
            xn, EDIM, w1t, ff1_b + l * FF, f1h, FF);

        // FF2 (K=1024, N=256) + residual accumulate into h (f32)
        gemm_wmma_kernel<FF, false, true, float><<<dim3(4, MT), 128, 0, stream>>>(
            f1h, FF, w2t, ff2_b + l * EDIM, h, EDIM);
    }

    // Classifier head on cls tokens
    head_kernel<<<1, 128, 0, stream>>>(h, fc_w, fc_b, (float*)d_out);
}
